// AttentionActorCritic_45045617000690
// MI455X (gfx1250) — matrix operand order repaired and checked by reading, not compiled, not run on hardware
//
#include <hip/hip_runtime.h>
#include <hip/hip_bf16.h>

typedef __attribute__((ext_vector_type(16))) _Float16 v16h;
typedef __attribute__((ext_vector_type(8)))  _Float16 v8h;
typedef __attribute__((ext_vector_type(8)))  float    v8f;

#define DEV __device__ __forceinline__

// ---------------- fragment table (packed f16 B-operands in d_ws) ----------------
enum {
  F_PW_QKV = 0,    // K32 x N96  -> 1x6
  F_PW_O   = 6,    // 32x32      -> 1x2
  F_PW_F1  = 8,
  F_PW_F2  = 10,
  F_PW_OUT = 12,   // K224 x N128 -> 7x8 = 56
  F_SW_QKV = 68,
  F_SW_O   = 74,
  F_SW_F1  = 76,
  F_SW_F2  = 78,
  F_SW_OUT = 80,   // K64 x N128  -> 2x8 = 16
  F_EW     = 96,   // K8->32 pad, N128 -> 1x8
  F_D1     = 104,  // K384 x N128 -> 12x8 = 96
  F_D2     = 200,  // K128 x N128 -> 4x8 = 32
  F_PI     = 232,  // K128 x N7->16 -> 4x1
  F_V      = 236,  // K128 x N1->16 -> 4x1
  F_TOTAL  = 240
};
// each fragment: 32 lanes * 16 halves = 512 halves = 1KB

// -------- LDS layout (bytes), 16 samples/tile, tokens channel-major --------
static constexpr int O_SIN  = 0;      // 16*30 f32         = 1920
static constexpr int O_XP   = 1920;   // 112*32 f16        = 7168
static constexpr int O_QKVP = 9088;   // 112*96 f16        = 21504
static constexpr int O_TMPP = 30592;  // 112*32 f16        = 7168
static constexpr int O_XS   = 37760;  // 32*32 f16         = 2048
static constexpr int O_QKVS = 39808;  // 32*96 f16         = 6144
static constexpr int O_TMPS = 45952;  // 32*32 f16         = 2048
static constexpr int O_PS   = 48000;  // 16*32 f16         = 1024
static constexpr int O_EMB  = 49024;  // 16*384 f16        = 12288
static constexpr int O_H1   = O_QKVP;         // reuse (branches done)
static constexpr int O_H2   = O_QKVP + 4096;  // 16*128 f16
static constexpr int SMEM_BYTES = 61312;

// ---------------- weight pack: fp32 [N,K] -> f16 WMMA B-fragments ----------------
// B 32x16 layout: N=lane%16; lane<16: halves = K0..7,K16..23 ; lane>=16: K8..15,K24..31
__global__ void pack_b(const float* __restrict__ W, _Float16* __restrict__ out,
                       int N, int K, int nNT, int fragBase) {
  int frag = blockIdx.x;
  int kt = frag / nNT, nt = frag % nNT;
  for (int i = threadIdx.x; i < 512; i += blockDim.x) {
    int l = i >> 4, h = i & 15;
    int n = nt * 16 + (l & 15);
    int k = kt * 32 + ((l < 16) ? 0 : 8) + ((h < 8) ? h : 8 + h);
    float v = (n < N && k < K) ? W[(size_t)n * K + k] : 0.f;
    out[(size_t)(fragBase + frag) * 512 + i] = (_Float16)v;
  }
}

// ---------------- WMMA helpers ----------------
DEV v8f vsplat(float x) {
  v8f z;
  #pragma unroll
  for (int i = 0; i < 8; ++i) z[i] = x;
  return z;
}

DEV v8f wmma16x16x32(v16h a, v16h b, v8f c) {
  return __builtin_amdgcn_wmma_f32_16x16x32_f16(false, a, false, b, (short)0, c, false, false);
}

// A 16x32 f16 layout: row M=lane%16; lane<16: halves = K0..7,K16..23 ; lane>=16: K8..15,K24..31
DEV v16h load_A(const _Float16* X, int rowBase, int colBase, int stride) {
  const int lane = threadIdx.x;
  const _Float16* pr = X + (size_t)(rowBase + (lane & 15)) * stride + colBase
                         + ((lane < 16) ? 0 : 8);
  v8h lo = *(const v8h*)pr;
  v8h hi = *(const v8h*)(pr + 16);
  v16h a;
  #pragma unroll
  for (int i = 0; i < 8; ++i) { a[i] = lo[i]; a[i + 8] = hi[i]; }
  return a;
}

DEV v16h load_B(const _Float16* F, int frag) {
  const _Float16* p = F + (size_t)frag * 512 + threadIdx.x * 16;
  return *(const v16h*)p;
}

// LayerNorm over 32 features on a D-fragment pair (cols 0..15 / 16..31), write f16
DEV void ln_store(_Float16* X, int rowBase, const float* y0, const float* y1,
                  const float* g, const float* b) {
  const int lane = threadIdx.x, col = lane & 15, r0 = (lane < 16) ? 0 : 8;
  float g0 = g[col], g1v = g[col + 16], b0 = b[col], b1v = b[col + 16];
  #pragma unroll
  for (int r = 0; r < 8; ++r) {
    float s = y0[r] + y1[r];
    float ss = y0[r] * y0[r] + y1[r] * y1[r];
    #pragma unroll
    for (int m = 1; m < 16; m <<= 1) { s += __shfl_xor(s, m); ss += __shfl_xor(ss, m); }
    float mu = s * (1.f / 32.f);
    float var = ss * (1.f / 32.f) - mu * mu;
    float inv = rsqrtf(var + 1e-5f);
    int row = rowBase + r0 + r;
    X[row * 32 + col]      = (_Float16)((y0[r] - mu) * inv * g0 + b0);
    X[row * 32 + col + 16] = (_Float16)((y1[r] - mu) * inv * g1v + b1v);
  }
}

// ---------------- one transformer encoder layer (post-norm, relu, E=32,H=4,D=8) ----------------
template <int CH>
DEV void encoder_block(_Float16* X, _Float16* QKV, _Float16* TMP,
                       const _Float16* F, int fQKV, int fO, int fF1, int fF2,
                       const float* qkv_b, const float* o_b,
                       const float* g1, const float* b1,
                       const float* f1b, const float* f2b,
                       const float* g2, const float* b2) {
  const int lane = threadIdx.x, col = lane & 15, r0 = (lane < 16) ? 0 : 8;

  // QKV projection: [CH*16,32] x [32,96], bias folded into C
  #pragma unroll
  for (int mt = 0; mt < CH; ++mt) {
    v16h a = load_A(X, mt * 16, 0, 32);
    #pragma unroll
    for (int nt = 0; nt < 6; ++nt) {
      v8f d = wmma16x16x32(a, load_B(F, fQKV + nt), vsplat(qkv_b[nt * 16 + col]));
      #pragma unroll
      for (int r = 0; r < 8; ++r)
        QKV[(mt * 16 + r0 + r) * 96 + nt * 16 + col] = (_Float16)d[r];
    }
  }
  __syncthreads();

  // attention: lane handles (s=lane%16, h=lane/16) and (s, h+2)
  {
    const float scale = 0.35355339059327373f; // 1/sqrt(8)
    const int s = lane & 15;
    #pragma unroll
    for (int hi = 0; hi < 2; ++hi) {
      const int h = (lane >> 4) + 2 * hi;
      float Kh[CH][8];
      v8h Vh[CH];
      #pragma unroll
      for (int c = 0; c < CH; ++c) {
        v8h kv = *(const v8h*)&QKV[(c * 16 + s) * 96 + 32 + h * 8];
        #pragma unroll
        for (int d = 0; d < 8; ++d) Kh[c][d] = (float)kv[d];
        Vh[c] = *(const v8h*)&QKV[(c * 16 + s) * 96 + 64 + h * 8];
      }
      #pragma unroll
      for (int ci = 0; ci < CH; ++ci) {
        v8h qv = *(const v8h*)&QKV[(ci * 16 + s) * 96 + h * 8];
        float q[8];
        #pragma unroll
        for (int d = 0; d < 8; ++d) q[d] = (float)qv[d];
        float sc[CH]; float mx = -1e30f;
        #pragma unroll
        for (int cj = 0; cj < CH; ++cj) {
          float acc = 0.f;
          #pragma unroll
          for (int d = 0; d < 8; ++d) acc += q[d] * Kh[cj][d];
          sc[cj] = acc * scale; mx = fmaxf(mx, sc[cj]);
        }
        float ssum = 0.f;
        #pragma unroll
        for (int cj = 0; cj < CH; ++cj) { float e = __expf(sc[cj] - mx); sc[cj] = e; ssum += e; }
        float inv = 1.f / ssum;
        v8h acc8;
        #pragma unroll
        for (int i = 0; i < 8; ++i) acc8[i] = (_Float16)0.f;
        #pragma unroll
        for (int cj = 0; cj < CH; ++cj) {
          _Float16 ah = (_Float16)(sc[cj] * inv);
          v8h aspl;
          #pragma unroll
          for (int i = 0; i < 8; ++i) aspl[i] = ah;
          acc8 = aspl * Vh[cj] + acc8;  // packed f16 fma
        }
        *(v8h*)&TMP[(ci * 16 + s) * 32 + h * 8] = acc8;
      }
    }
  }
  __syncthreads();

  // o-proj (+bias in C) + residual + LN1
  #pragma unroll
  for (int mt = 0; mt < CH; ++mt) {
    v16h a = load_A(TMP, mt * 16, 0, 32);
    v8f d0 = wmma16x16x32(a, load_B(F, fO + 0), vsplat(o_b[col]));
    v8f d1 = wmma16x16x32(a, load_B(F, fO + 1), vsplat(o_b[col + 16]));
    float y0[8], y1[8];
    #pragma unroll
    for (int r = 0; r < 8; ++r) {
      int row = mt * 16 + r0 + r;
      y0[r] = d0[r] + (float)X[row * 32 + col];
      y1[r] = d1[r] + (float)X[row * 32 + col + 16];
    }
    ln_store(X, mt * 16, y0, y1, g1, b1);
  }
  __syncthreads();

  // FF1 (+bias in C, relu)
  #pragma unroll
  for (int mt = 0; mt < CH; ++mt) {
    v16h a = load_A(X, mt * 16, 0, 32);
    v8f d0 = wmma16x16x32(a, load_B(F, fF1 + 0), vsplat(f1b[col]));
    v8f d1 = wmma16x16x32(a, load_B(F, fF1 + 1), vsplat(f1b[col + 16]));
    #pragma unroll
    for (int r = 0; r < 8; ++r) {
      int row = mt * 16 + r0 + r;
      TMP[row * 32 + col]      = (_Float16)fmaxf(d0[r], 0.f);
      TMP[row * 32 + col + 16] = (_Float16)fmaxf(d1[r], 0.f);
    }
  }
  __syncthreads();

  // FF2 (+bias in C) + residual + LN2
  #pragma unroll
  for (int mt = 0; mt < CH; ++mt) {
    v16h a = load_A(TMP, mt * 16, 0, 32);
    v8f d0 = wmma16x16x32(a, load_B(F, fF2 + 0), vsplat(f2b[col]));
    v8f d1 = wmma16x16x32(a, load_B(F, fF2 + 1), vsplat(f2b[col + 16]));
    float y0[8], y1[8];
    #pragma unroll
    for (int r = 0; r < 8; ++r) {
      int row = mt * 16 + r0 + r;
      y0[r] = d0[r] + (float)X[row * 32 + col];
      y1[r] = d1[r] + (float)X[row * 32 + col + 16];
    }
    ln_store(X, mt * 16, y0, y1, g2, b2);
  }
  __syncthreads();
}

// dense -> 128 outputs (bias in C), relu, store f16 into `out` at column offset
template <int KT>
DEV void dense128_relu(const _Float16* Xin, int stride, int rowStep, int colStep,
                       const _Float16* F, int fBase, const float* bias,
                       _Float16* out, int outStride, int outColBase) {
  const int lane = threadIdx.x, col = lane & 15, r0 = (lane < 16) ? 0 : 8;
  v8f acc[8];
  #pragma unroll
  for (int nt = 0; nt < 8; ++nt) acc[nt] = vsplat(bias[nt * 16 + col]);
  #pragma unroll
  for (int kt = 0; kt < KT; ++kt) {
    v16h a = load_A(Xin, kt * rowStep, kt * colStep, stride);
    #pragma unroll
    for (int nt = 0; nt < 8; ++nt)
      acc[nt] = wmma16x16x32(a, load_B(F, fBase + kt * 8 + nt), acc[nt]);
  }
  #pragma unroll
  for (int nt = 0; nt < 8; ++nt) {
    #pragma unroll
    for (int r = 0; r < 8; ++r)
      out[(r0 + r) * outStride + outColBase + nt * 16 + col] =
          (_Float16)fmaxf(acc[nt][r], 0.f);
  }
}

// ---------------- kernel params ----------------
struct KParams {
  const float *state;
  const float *pw_in_w, *pw_in_b, *pw_qkv_b, *pw_o_b, *pw_g1, *pw_b1,
              *pw_f1b, *pw_f2b, *pw_g2, *pw_b2, *pw_out_b;
  const float *sw_in_w, *sw_in_b, *sw_qkv_b, *sw_o_b, *sw_g1, *sw_b1,
              *sw_f1b, *sw_f2b, *sw_g2, *sw_b2, *sw_out_b;
  const float *e_b, *d1_b, *d2_b, *pi_b, *v_b;
  const _Float16* F;
  float *outPi, *outV;
};

__global__ __launch_bounds__(32) void aac_fwd(KParams p) {
  extern __shared__ __align__(16) char smem[];
  float*    SIN  = (float*)(smem + O_SIN);
  _Float16* XP   = (_Float16*)(smem + O_XP);
  _Float16* QKVP = (_Float16*)(smem + O_QKVP);
  _Float16* TMPP = (_Float16*)(smem + O_TMPP);
  _Float16* XS   = (_Float16*)(smem + O_XS);
  _Float16* QKVS = (_Float16*)(smem + O_QKVS);
  _Float16* TMPS = (_Float16*)(smem + O_TMPS);
  _Float16* PS   = (_Float16*)(smem + O_PS);
  _Float16* EMB  = (_Float16*)(smem + O_EMB);
  _Float16* H1   = (_Float16*)(smem + O_H1);
  _Float16* H2   = (_Float16*)(smem + O_H2);

  const int lane = threadIdx.x;
  const int tileBase = blockIdx.x * 16;

  // stage input rows [16 x 30] fp32
  const float* st = p.state + (size_t)tileBase * 30;
  for (int i = lane; i < 16 * 30; i += 32) SIN[i] = st[i];
  __syncthreads();

  // token embeddings (tiny K: VALU). channel-major rows: row = c*16 + s
  for (int i = lane; i < 112 * 32; i += 32) {
    int row = i >> 5, c = row >> 4, s = row & 15, e = i & 31;
    float f0 = SIN[s * 30 + c * 2 + 0];
    float f1 = SIN[s * 30 + c * 2 + 1];
    XP[i] = (_Float16)(p.pw_in_w[e * 2] * f0 + p.pw_in_w[e * 2 + 1] * f1 + p.pw_in_b[e]);
  }
  for (int i = lane; i < 32 * 32; i += 32) {
    int row = i >> 5, c = row >> 4, s = row & 15, e = i & 31;
    const float* f = SIN + s * 30 + 14 + c * 4;
    XS[i] = (_Float16)(p.sw_in_w[e * 4] * f[0] + p.sw_in_w[e * 4 + 1] * f[1] +
                       p.sw_in_w[e * 4 + 2] * f[2] + p.sw_in_w[e * 4 + 3] * f[3] +
                       p.sw_in_b[e]);
  }
  for (int i = lane; i < 16 * 32; i += 32) {
    int s = i >> 5, e = i & 31;
    PS[i] = (_Float16)((e < 8) ? SIN[s * 30 + 22 + e] : 0.f);
  }
  __syncthreads();

  // power branch
  encoder_block<7>(XP, QKVP, TMPP, p.F, F_PW_QKV, F_PW_O, F_PW_F1, F_PW_F2,
                   p.pw_qkv_b, p.pw_o_b, p.pw_g1, p.pw_b1, p.pw_f1b, p.pw_f2b,
                   p.pw_g2, p.pw_b2);
  dense128_relu<7>(XP, 32, 16, 0, p.F, F_PW_OUT, p.pw_out_b, EMB, 384, 0);
  __syncthreads();

  // sword branch
  encoder_block<2>(XS, QKVS, TMPS, p.F, F_SW_QKV, F_SW_O, F_SW_F1, F_SW_F2,
                   p.sw_qkv_b, p.sw_o_b, p.sw_g1, p.sw_b1, p.sw_f1b, p.sw_f2b,
                   p.sw_g2, p.sw_b2);
  dense128_relu<2>(XS, 32, 16, 0, p.F, F_SW_OUT, p.sw_out_b, EMB, 384, 128);
  // state embed (K padded 8->32 with zeros on both operands)
  dense128_relu<1>(PS, 32, 0, 0, p.F, F_EW, p.e_b, EMB, 384, 256);
  __syncthreads();

  // trunk
  dense128_relu<12>(EMB, 384, 0, 32, p.F, F_D1, p.d1_b, H1, 128, 0);
  __syncthreads();
  dense128_relu<4>(H1, 128, 0, 32, p.F, F_D2, p.d2_b, H2, 128, 0);
  __syncthreads();

  const int col = lane & 15, r0 = (lane < 16) ? 0 : 8;

  // policy head + softmax over 7 actions
  {
    bool ok = col < 7;
    float pb = p.pi_b[ok ? col : 0];
    v8f acc = vsplat(pb);
    #pragma unroll
    for (int kt = 0; kt < 4; ++kt)
      acc = wmma16x16x32(load_A(H2, 0, kt * 32, 128), load_B(p.F, F_PI + kt), acc);
    #pragma unroll
    for (int r = 0; r < 8; ++r) {
      float x = acc[r];
      float xm = ok ? x : -1e30f;
      #pragma unroll
      for (int m = 1; m < 16; m <<= 1) xm = fmaxf(xm, __shfl_xor(xm, m));
      float e = ok ? __expf(x - xm) : 0.f;
      float s = e;
      #pragma unroll
      for (int m = 1; m < 16; m <<= 1) s += __shfl_xor(s, m);
      if (ok) p.outPi[(size_t)(tileBase + r0 + r) * 7 + col] = e / s;
    }
  }
  // value head
  {
    v8f acc = vsplat(p.v_b[0]);
    #pragma unroll
    for (int kt = 0; kt < 4; ++kt)
      acc = wmma16x16x32(load_A(H2, 0, kt * 32, 128), load_B(p.F, F_V + kt), acc);
    if (col == 0) {
      #pragma unroll
      for (int r = 0; r < 8; ++r)
        p.outV[tileBase + r0 + r] = acc[r];
    }
  }
}

// ---------------- host entry ----------------
extern "C" void kernel_launch(void* const* d_in, const int* in_sizes, int n_in,
                              void* d_out, int out_size, void* d_ws, size_t ws_size,
                              hipStream_t stream) {
  (void)out_size; (void)ws_size;
  const float* IN[43];
  for (int i = 0; i < 43 && i < n_in; ++i) IN[i] = (const float*)d_in[i];
  _Float16* frags = (_Float16*)d_ws;

  auto pack = [&](const float* W, int N, int K, int base) {
    int nNT = (N + 15) / 16, nKT = (K + 31) / 32;
    pack_b<<<dim3(nKT * nNT), dim3(64), 0, stream>>>(W, frags, N, K, nNT, base);
  };
  pack(IN[3], 96, 32, F_PW_QKV);
  pack(IN[5], 32, 32, F_PW_O);
  pack(IN[9], 32, 32, F_PW_F1);
  pack(IN[11], 32, 32, F_PW_F2);
  pack(IN[15], 128, 224, F_PW_OUT);
  pack(IN[19], 96, 32, F_SW_QKV);
  pack(IN[21], 32, 32, F_SW_O);
  pack(IN[25], 32, 32, F_SW_F1);
  pack(IN[27], 32, 32, F_SW_F2);
  pack(IN[31], 128, 64, F_SW_OUT);
  pack(IN[33], 128, 8, F_EW);
  pack(IN[35], 128, 384, F_D1);
  pack(IN[37], 128, 128, F_D2);
  pack(IN[39], 7, 128, F_PI);
  pack(IN[41], 1, 128, F_V);

  int Btot = in_sizes[0] / 30;

  KParams p{};
  p.state = IN[0];
  p.pw_in_w = IN[1];  p.pw_in_b = IN[2];
  p.pw_qkv_b = IN[4]; p.pw_o_b = IN[6];
  p.pw_g1 = IN[7];    p.pw_b1 = IN[8];
  p.pw_f1b = IN[10];  p.pw_f2b = IN[12];
  p.pw_g2 = IN[13];   p.pw_b2 = IN[14];
  p.pw_out_b = IN[16];
  p.sw_in_w = IN[17]; p.sw_in_b = IN[18];
  p.sw_qkv_b = IN[20]; p.sw_o_b = IN[22];
  p.sw_g1 = IN[23];   p.sw_b1 = IN[24];
  p.sw_f1b = IN[26];  p.sw_f2b = IN[28];
  p.sw_g2 = IN[29];   p.sw_b2 = IN[30];
  p.sw_out_b = IN[32];
  p.e_b = IN[34]; p.d1_b = IN[36]; p.d2_b = IN[38];
  p.pi_b = IN[40]; p.v_b = IN[42];
  p.F = frags;
  p.outPi = (float*)d_out;
  p.outV = (float*)d_out + (size_t)Btot * 7;

  aac_fwd<<<dim3(Btot / 16), dim3(32), SMEM_BYTES, stream>>>(p);
}
